// SovereignPhiAttention_9363028705344
// MI455X (gfx1250) — matrix operand order repaired and checked by reading, not compiled, not run on hardware
//
#include <hip/hip_runtime.h>
#include <cstdint>
#include <cstddef>

// ---------------------------------------------------------------------------
// Types for CDNA5 WMMA (wave32): V_WMMA_F32_16X16X32_BF16
// ---------------------------------------------------------------------------
typedef __attribute__((ext_vector_type(16))) __bf16 bf16x16;
typedef __attribute__((ext_vector_type(8)))  __bf16 bf16x8;
typedef __attribute__((ext_vector_type(8)))  float  f32x8;

union ABFrag { bf16x16 v; bf16x8 h[2]; };

#define HIDDEN 2048
#define SEQ    2048
#define BATCH  2
#define NH     32
#define NKV    8
#define HD     64
#define QKV_N  3072          // (32 + 2*8) * 64
#define ROWS   (BATCH*SEQ)   // 4096

// float -> bf16, round-to-nearest-even
__device__ __forceinline__ __bf16 f2bf(float f) {
  unsigned u = __builtin_bit_cast(unsigned, f);
  u += 0x7fffu + ((u >> 16) & 1u);
  unsigned short s = (unsigned short)(u >> 16);
  return __builtin_bit_cast(__bf16, s);
}

__device__ __forceinline__ f32x8 zero8() {
  f32x8 z;
#pragma unroll
  for (int i = 0; i < 8; ++i) z[i] = 0.0f;
  return z;
}

__device__ __forceinline__ f32x8 wmma_bf16(bf16x16 a, bf16x16 b, f32x8 c) {
  // (neg_a, A, neg_b, B, c_mod, C, reuse_a, reuse_b)
  return __builtin_amdgcn_wmma_f32_16x16x32_bf16(false, a, false, b, (short)0, c,
                                                 false, false);
}

// Wave-local LDS fence: DS ops are in-order per wave; this waits for outstanding
// DS traffic and pins compiler/scheduler ordering around the LDS round-trip.
__device__ __forceinline__ void lds_wave_fence() {
  asm volatile("s_wait_dscnt 0" ::: "memory");
}

// ---------------------------------------------------------------------------
// Prep kernels
// ---------------------------------------------------------------------------
__global__ void k_f32_to_bf16(const float* __restrict__ in,
                              __bf16* __restrict__ out, int n) {
  int i = blockIdx.x * 256 + threadIdx.x;
  if (i < n) out[i] = f2bf(in[i]);
}

// in: f32 [R][C] row-major  ->  out: bf16 [C][R] row-major (transposed)
__global__ void k_transpose_to_bf16(const float* __restrict__ in,
                                    __bf16* __restrict__ out, int R, int C) {
  __shared__ float tile[32][33];
  int c0 = blockIdx.x * 32, r0 = blockIdx.y * 32;
  int x = c0 + threadIdx.x;
  for (int i = threadIdx.y; i < 32; i += 8) {
    int y = r0 + i;
    if (y < R && x < C) tile[i][threadIdx.x] = in[(size_t)y * C + x];
  }
  __syncthreads();
  int ox = r0 + threadIdx.x;
  for (int i = threadIdx.y; i < 32; i += 8) {
    int oc = c0 + i;
    if (oc < C && ox < R) out[(size_t)oc * R + ox] = f2bf(tile[threadIdx.x][i]);
  }
}

// RoPE table: ct/st [SEQ][32], angle = pos * theta^(-j/32)
__global__ void k_rope_table(const int* __restrict__ pos,
                             float* __restrict__ ct, float* __restrict__ st) {
  int i = blockIdx.x * 256 + threadIdx.x;
  if (i >= SEQ * 32) return;
  int s = i >> 5, j = i & 31;
  float invf = __expf(-0.28782313f * (float)j);  // ln(10000)/32 = 0.28782313
  float ang = (float)pos[s] * invf;
  ct[i] = cosf(ang);
  st[i] = sinf(ang);
}

// ---------------------------------------------------------------------------
// Kernel 1: qkv = x @ Wqkv, fused RoPE on Q/K, scatter to attention layouts
//   A-frags from x_bf16 [row][k]; B-frags from WqkvT [n][k] (contig 32B/lane)
//   wave tile: 32 (M) x 64 (N == one head)
// ---------------------------------------------------------------------------
__global__ void __launch_bounds__(256, 2)
k_qkv_rope(const __bf16* __restrict__ xb, const __bf16* __restrict__ wT,
           const float* __restrict__ ct, const float* __restrict__ st,
           __bf16* __restrict__ qb, __bf16* __restrict__ kb,
           __bf16* __restrict__ vtb) {
  const int lane = threadIdx.x & 31;
  const int widx = threadIdx.x >> 5;
  const int l15 = lane & 15;
  const int hi  = lane >> 4;
  const int wg  = blockIdx.x * 8 + widx;
  const int slab = wg % 48;         // 48 head-slabs of 64 cols
  const int m0   = (wg / 48) * 32;  // 32 rows per wave
  const int c0n  = slab * 64;

  f32x8 acc[2][4];
#pragma unroll
  for (int a = 0; a < 2; ++a)
#pragma unroll
    for (int j = 0; j < 4; ++j) acc[a][j] = zero8();

  const size_t rowA0 = (size_t)(m0 + l15) * HIDDEN;
  const size_t rowA1 = (size_t)(m0 + 16 + l15) * HIDDEN;

  for (int k0 = 0; k0 < HIDDEN; k0 += 32) {
    ABFrag a0, a1, bfr[4];
    const int ka = k0 + hi * 16;   // A: linear K layout per lane-half
    a0.h[0] = *(const bf16x8*)(xb + rowA0 + ka);
    a0.h[1] = *(const bf16x8*)(xb + rowA0 + ka + 8);
    a1.h[0] = *(const bf16x8*)(xb + rowA1 + ka);
    a1.h[1] = *(const bf16x8*)(xb + rowA1 + ka + 8);
    __builtin_prefetch(xb + rowA0 + ka + 128, 0, 3);  // WGP-scope prefetch
    const int kbq = k0 + hi * 16;  // B: linear K layout per lane-half
#pragma unroll
    for (int j = 0; j < 4; ++j)
      bfr[j].v = *(const bf16x16*)(wT + (size_t)(c0n + 16 * j + l15) * HIDDEN + kbq);
#pragma unroll
    for (int j = 0; j < 4; ++j) {
      acc[0][j] = wmma_bf16(a0.v, bfr[j].v, acc[0][j]);
      acc[1][j] = wmma_bf16(a1.v, bfr[j].v, acc[1][j]);
    }
  }

  // Epilogue: RoPE (Q heads 0..31, K heads 32..39) + scatter stores (bf16)
#pragma unroll
  for (int mt = 0; mt < 2; ++mt) {
#pragma unroll
    for (int i = 0; i < 8; ++i) {
      const int row = m0 + 16 * mt + i + 8 * hi;   // global token row
      const int b = row >> 11;                     // / SEQ
      const int s = row & (SEQ - 1);
      float c0 = acc[mt][0][i], c1 = acc[mt][1][i];
      float c2 = acc[mt][2][i], c3 = acc[mt][3][i];
      if (slab < 40) {  // RoPE applies to Q and K
        const float cA = ct[s * 32 + l15];
        const float sA = st[s * 32 + l15];
        const float cB = ct[s * 32 + 16 + l15];
        const float sB = st[s * 32 + 16 + l15];
        float n0 = c0 * cA - c2 * sA;   // d < 32 : q*cos - q[d+32]*sin
        float n1 = c1 * cB - c3 * sB;
        float n2 = c2 * cA + c0 * sA;   // d >= 32: q*cos + q[d-32]*sin
        float n3 = c3 * cB + c1 * sB;
        c0 = n0; c1 = n1; c2 = n2; c3 = n3;
      }
      const float cv[4] = {c0, c1, c2, c3};
#pragma unroll
      for (int j = 0; j < 4; ++j) {
        const int d = 16 * j + l15;
        if (slab < 32) {            // Q -> [b][h][s][d]
          qb[((size_t)(b * NH + slab) * SEQ + s) * HD + d] = f2bf(cv[j]);
        } else if (slab < 40) {     // K -> [b][kv][s][d]
          kb[((size_t)(b * NKV + (slab - 32)) * SEQ + s) * HD + d] = f2bf(cv[j]);
        } else {                    // V -> transposed [b][kv][d][s]
          vtb[((size_t)(b * NKV + (slab - 40)) * HD + d) * SEQ + s] = f2bf(cv[j]);
        }
      }
    }
  }
}

// ---------------------------------------------------------------------------
// Kernel 2: flash-style attention. One wave per 16-query tile; 64-key chunks.
// Waves are fully independent: LDS P-slabs are per-wave, synced with a
// wave-local s_wait_dscnt instead of block barriers.
// ---------------------------------------------------------------------------
__global__ void __launch_bounds__(256, 2)
k_attention(const __bf16* __restrict__ qb, const __bf16* __restrict__ kb,
            const __bf16* __restrict__ vtb, const float* __restrict__ mask,
            __bf16* __restrict__ ob) {
  __shared__ __align__(128) __bf16 Plds[8 * 16 * 64];  // 16KB, per-wave slabs
  const int lane = threadIdx.x & 31;
  const int widx = threadIdx.x >> 5;
  const int l15 = lane & 15;
  const int hi  = lane >> 4;
  int wg = blockIdx.x * 8 + widx;
  const int qt = wg & 127;  wg >>= 7;          // 128 q-tiles
  const int h  = wg & 31;   const int b = wg >> 5;
  const int kvh = h >> 2;                       // GQA: 4 q-heads per kv-head
  const int s0 = qt * 16;

  // Q A-fragments (16 x 64 -> two 16x32 frags)
  ABFrag aQ[2];
  {
    const __bf16* qrow = qb + (size_t)((b * NH + h) * SEQ + s0 + l15) * HD;
    const int d0 = hi * 16;
    aQ[0].h[0] = *(const bf16x8*)(qrow + d0);
    aQ[0].h[1] = *(const bf16x8*)(qrow + d0 + 8);
    aQ[1].h[0] = *(const bf16x8*)(qrow + 32 + d0);
    aQ[1].h[1] = *(const bf16x8*)(qrow + 32 + d0 + 8);
  }
  const __bf16* kbase = kb  + (size_t)(b * NKV + kvh) * SEQ * HD;
  const __bf16* vbase = vtb + (size_t)(b * NKV + kvh) * HD * SEQ;

  f32x8 accO[4];
#pragma unroll
  for (int t = 0; t < 4; ++t) accO[t] = zero8();
  float m_[8], l_[8];
#pragma unroll
  for (int i = 0; i < 8; ++i) { m_[i] = -1e30f; l_[i] = 0.0f; }

  __bf16* pw = Plds + widx * 1024;   // this wave's 16x64 P slab
  const int qrow0 = s0 + hi * 8;     // C-frag element i -> query row qrow0+i

  for (int kc = 0; kc < SEQ; kc += 64) {
    // ---- S = Q @ K^T for 16q x 64k -------------------------------------
    f32x8 cS[4];
#pragma unroll
    for (int j = 0; j < 4; ++j) {
      const __bf16* krow = kbase + (size_t)(kc + 16 * j + l15) * HD;
      ABFrag bk0, bk1;
      bk0.v = *(const bf16x16*)(krow + hi * 16);        // d 0..31
      bk1.v = *(const bf16x16*)(krow + 32 + hi * 16);   // d 32..63
      f32x8 c = zero8();
      c = wmma_bf16(aQ[0].v, bk0.v, c);
      c = wmma_bf16(aQ[1].v, bk1.v, c);
      cS[j] = c;
    }
    __builtin_prefetch(kbase + (size_t)(kc + 64 + l15) * HD, 0, 3);
    __builtin_prefetch(vbase + (size_t)l15 * SEQ + kc + 64, 0, 3);

    // ---- scale + mask, online softmax ----------------------------------
    float sc[4][8], rm[8];
#pragma unroll
    for (int i = 0; i < 8; ++i) rm[i] = -1e30f;
#pragma unroll
    for (int j = 0; j < 4; ++j)
#pragma unroll
      for (int i = 0; i < 8; ++i) {
        float v = cS[j][i] * 0.125f +
                  mask[(size_t)(qrow0 + i) * SEQ + kc + 16 * j + l15];
        sc[j][i] = v;
        rm[i] = fmaxf(rm[i], v);
      }
#pragma unroll
    for (int i = 0; i < 8; ++i) {
      float r = rm[i];
      r = fmaxf(r, __shfl_xor(r, 1));
      r = fmaxf(r, __shfl_xor(r, 2));
      r = fmaxf(r, __shfl_xor(r, 4));
      r = fmaxf(r, __shfl_xor(r, 8));   // stays within the 16-lane half
      float nm = fmaxf(m_[i], r);
      float alpha = __expf(m_[i] - nm);
      m_[i] = nm;
      l_[i] *= alpha;
#pragma unroll
      for (int t = 0; t < 4; ++t) accO[t][i] *= alpha;
    }
    // ---- P = exp(S - m), row sums, transpose via LDS -------------------
    float rs[8];
#pragma unroll
    for (int i = 0; i < 8; ++i) rs[i] = 0.0f;
#pragma unroll
    for (int j = 0; j < 4; ++j)
#pragma unroll
      for (int i = 0; i < 8; ++i) {
        float p = __expf(sc[j][i] - m_[i]);
        rs[i] += p;
        pw[(i + 8 * hi) * 64 + 16 * j + l15] = f2bf(p);
      }
#pragma unroll
    for (int i = 0; i < 8; ++i) {
      float r = rs[i];
      r += __shfl_xor(r, 1);
      r += __shfl_xor(r, 2);
      r += __shfl_xor(r, 4);
      r += __shfl_xor(r, 8);
      l_[i] += r;
    }
    lds_wave_fence();  // writes -> reads lane exchange (wave-local)

    // ---- read P back in A-fragment layout ------------------------------
    ABFrag aP0, aP1;
    {
      const __bf16* prow = pw + l15 * 64;
      const int k8 = hi * 16;
      aP0.h[0] = *(const bf16x8*)(prow + k8);
      aP0.h[1] = *(const bf16x8*)(prow + k8 + 8);
      aP1.h[0] = *(const bf16x8*)(prow + 32 + k8);
      aP1.h[1] = *(const bf16x8*)(prow + 32 + k8 + 8);
    }
    // ---- O += P @ V -----------------------------------------------------
#pragma unroll
    for (int t = 0; t < 4; ++t) {
      const __bf16* vrow = vbase + (size_t)(16 * t + l15) * SEQ + kc;
      ABFrag bv0, bv1;
      bv0.v = *(const bf16x16*)(vrow + hi * 16);        // keys 0..31 of chunk
      bv1.v = *(const bf16x16*)(vrow + 32 + hi * 16);   // keys 32..63
      accO[t] = wmma_bf16(aP0.v, bv0.v, accO[t]);
      accO[t] = wmma_bf16(aP1.v, bv1.v, accO[t]);
    }
    lds_wave_fence();  // reads done before next iteration's writes
  }

  // ---- normalize and store O (bf16) as [b*S][H*D] -----------------------
  float inv[8];
#pragma unroll
  for (int i = 0; i < 8; ++i) inv[i] = 1.0f / l_[i];
#pragma unroll
  for (int t = 0; t < 4; ++t)
#pragma unroll
    for (int i = 0; i < 8; ++i) {
      const size_t row = (size_t)b * SEQ + s0 + i + 8 * hi;
      ob[row * HIDDEN + h * HD + 16 * t + l15] = f2bf(accO[t][i] * inv[i]);
    }
}

// ---------------------------------------------------------------------------
// Kernel 3: out = attn @ Wo   (A: bf16 [4096][2048], B: WoT bf16 [n][k])
// ---------------------------------------------------------------------------
__global__ void __launch_bounds__(256, 2)
k_out_gemm(const __bf16* __restrict__ ab, const __bf16* __restrict__ woT,
           float* __restrict__ out) {
  const int lane = threadIdx.x & 31;
  const int widx = threadIdx.x >> 5;
  const int l15 = lane & 15;
  const int hi  = lane >> 4;
  const int wg  = blockIdx.x * 8 + widx;
  const int slab = wg & 31;          // 32 slabs of 64 cols
  const int m0   = (wg >> 5) * 32;
  const int c0n  = slab * 64;

  f32x8 acc[2][4];
#pragma unroll
  for (int a = 0; a < 2; ++a)
#pragma unroll
    for (int j = 0; j < 4; ++j) acc[a][j] = zero8();

  const size_t rowA0 = (size_t)(m0 + l15) * HIDDEN;
  const size_t rowA1 = (size_t)(m0 + 16 + l15) * HIDDEN;

  for (int k0 = 0; k0 < HIDDEN; k0 += 32) {
    ABFrag a0, a1, bfr[4];
    const int ka = k0 + hi * 16;
    a0.h[0] = *(const bf16x8*)(ab + rowA0 + ka);
    a0.h[1] = *(const bf16x8*)(ab + rowA0 + ka + 8);
    a1.h[0] = *(const bf16x8*)(ab + rowA1 + ka);
    a1.h[1] = *(const bf16x8*)(ab + rowA1 + ka + 8);
    __builtin_prefetch(ab + rowA0 + ka + 128, 0, 3);  // WGP-scope prefetch
    const int kbq = k0 + hi * 16;
#pragma unroll
    for (int j = 0; j < 4; ++j)
      bfr[j].v = *(const bf16x16*)(woT + (size_t)(c0n + 16 * j + l15) * HIDDEN + kbq);
#pragma unroll
    for (int j = 0; j < 4; ++j) {
      acc[0][j] = wmma_bf16(a0.v, bfr[j].v, acc[0][j]);
      acc[1][j] = wmma_bf16(a1.v, bfr[j].v, acc[1][j]);
    }
  }
#pragma unroll
  for (int mt = 0; mt < 2; ++mt)
#pragma unroll
    for (int j = 0; j < 4; ++j)
#pragma unroll
      for (int i = 0; i < 8; ++i) {
        const size_t row = (size_t)m0 + 16 * mt + i + 8 * hi;
        out[row * HIDDEN + c0n + 16 * j + l15] = acc[mt][j][i];
      }
}

// ---------------------------------------------------------------------------
// Launch
// ---------------------------------------------------------------------------
extern "C" void kernel_launch(void* const* d_in, const int* in_sizes, int n_in,
                              void* d_out, int out_size, void* d_ws,
                              size_t ws_size, hipStream_t stream) {
  (void)in_sizes; (void)n_in; (void)out_size; (void)ws_size;
  const float* x    = (const float*)d_in[0];
  const int*   pos  = (const int*)d_in[1];
  const float* mask = (const float*)d_in[2];
  const float* wqkv = (const float*)d_in[3];
  const float* wo   = (const float*)d_in[4];

  char* ws = (char*)d_ws;
  __bf16* xb   = (__bf16*)(ws);                  // 16 MB  x in bf16
  __bf16* wqT  = (__bf16*)(ws + 16777216);       // 12 MB  Wqkv^T bf16 [3072][2048]
  __bf16* woT  = (__bf16*)(ws + 29360128);       //  8 MB  Wo^T   bf16 [2048][2048]
  float*  ct   = (float*) (ws + 37748736);       // 256 KB cos table
  float*  st   = (float*) (ws + 38010880);       // 256 KB sin table
  __bf16* qbuf = (__bf16*)(ws + 38273024);       // 16 MB  Q roped  [b][h][s][d]
  __bf16* kbuf = (__bf16*)(ws + 55050240);       //  4 MB  K roped  [b][kv][s][d]
  __bf16* vtb  = (__bf16*)(ws + 59244544);       //  4 MB  V^T      [b][kv][d][s]
  __bf16* abuf = (__bf16*)(ws + 63438848);       // 16 MB  attn out [b*s][h*d]

  k_f32_to_bf16<<<(ROWS * HIDDEN + 255) / 256, 256, 0, stream>>>(x, xb,
                                                                 ROWS * HIDDEN);
  k_transpose_to_bf16<<<dim3(QKV_N / 32, HIDDEN / 32), dim3(32, 8), 0, stream>>>(
      wqkv, wqT, HIDDEN, QKV_N);
  k_transpose_to_bf16<<<dim3(HIDDEN / 32, HIDDEN / 32), dim3(32, 8), 0, stream>>>(
      wo, woT, HIDDEN, HIDDEN);
  k_rope_table<<<(SEQ * 32 + 255) / 256, 256, 0, stream>>>(pos, ct, st);

  k_qkv_rope<<<(128 * 48) / 8, 256, 0, stream>>>(xb, wqT, ct, st, qbuf, kbuf, vtb);
  k_attention<<<(BATCH * NH * (SEQ / 16)) / 8, 256, 0, stream>>>(qbuf, kbuf, vtb,
                                                                 mask, abuf);
  k_out_gemm<<<(128 * 32) / 8, 256, 0, stream>>>(abuf, woT, (float*)d_out);
}
